// MultiHeadedAttention_89799176224812
// MI455X (gfx1250) — compile-verified, matrix operand order checked by reading, not run on hardware
//
#include <hip/hip_runtime.h>
#include <hip/hip_bf16.h>
#include <stdint.h>

// MI455X / gfx1250, wave32. Matmuls via v_wmma_f32_16x16x32_bf16;
// tile staging via global_load_async_to_lds_b128 (ASYNCcnt) when available.

typedef __attribute__((ext_vector_type(16))) __bf16 v16bf;
typedef __attribute__((ext_vector_type(8)))  float  v8f;
typedef __attribute__((ext_vector_type(4)))  int    v4i_;

#define S_LEN  2048
#define DMODEL 1024
#define NHEAD  16
#define DHEAD  64
#define BATCH  2

#define AS1 __attribute__((address_space(1)))
#define AS3 __attribute__((address_space(3)))

#if defined(__has_builtin)
#if __has_builtin(__builtin_amdgcn_global_load_async_to_lds_b128)
#define HAVE_ASYNC_LDS 1
#endif
#if __has_builtin(__builtin_amdgcn_s_wait_asynccnt)
#define HAVE_ASYNC_WAIT 1
#endif
#endif

#ifdef HAVE_ASYNC_LDS
__device__ __forceinline__ void async_copy_b128(const void* g, void* l) {
  // (global src as v4i*, LDS dst as v4i*, imm offset, imm cpol).
  // LDS generic pointer's low 32 bits are the group-segment offset.
  __builtin_amdgcn_global_load_async_to_lds_b128(
      (AS1 v4i_*)(uintptr_t)g, (AS3 v4i_*)(uint32_t)(uintptr_t)l, 0, 0);
}
__device__ __forceinline__ void wait_async0() {
#ifdef HAVE_ASYNC_WAIT
  __builtin_amdgcn_s_wait_asynccnt(0);
#else
  asm volatile("s_wait_asynccnt 0" ::: "memory");
#endif
}
#endif

// Load a 16x32 bf16 WMMA fragment (A-style: lane l16 = row, kmap pairs
// {0..7,16..23} + 8*half + koff). B-fragments use the same loader on a
// transposed LDS tile (lane = column of B).
__device__ __forceinline__ v16bf load_frag_b(const __bf16* base, int stride,
                                             int row, int half, int koff) {
  v16bf f;
  const __bf16* rp = base + row * stride + 8 * half + koff;
#pragma unroll
  for (int j = 0; j < 8; ++j) {
    int kk = (j < 4) ? (2 * j) : (16 + 2 * (j - 4));
    f[2 * j]     = rp[kk];
    f[2 * j + 1] = rp[kk + 1];
  }
  return f;
}

// ---------------------------------------------------------------------------
// Generic 4096x1024x1024 GEMM: out = (A @ Bw + bias) * scale
// A_F32: A operand is f32 (converted to bf16 on LDS stage) else bf16.
// HEADED: write bf16 output in [B, H, S, DH] layout, else f32 row-major.
// Block: 256 threads (8 waves), tile 128(M) x 128(N), K staged 64 deep.
// ---------------------------------------------------------------------------
template <bool A_F32, bool HEADED>
__global__ __launch_bounds__(256) void gemm_wmma_k(
    const void* __restrict__ Av, const float* __restrict__ Bw,
    const float* __restrict__ bias, void* __restrict__ Ov, float scale) {
  constexpr int N = DMODEL, K = DMODEL;
  constexpr int LD = 66;  // 33-dword row stride: conflict-free
  __shared__ __bf16 As[128 * LD];
  __shared__ __bf16 Bs[128 * LD];

  const int tid  = threadIdx.x;
  const int lane = tid & 31, wave = tid >> 5;
  const int l16 = lane & 15, half = lane >> 4;
  const int wm = wave & 3;   // 32-row subtile
  const int wn = wave >> 2;  // 64-col subtile
  const int m0 = blockIdx.y * 128;
  const int n0 = blockIdx.x * 128;

  v8f acc[2][4] = {};

  for (int k0 = 0; k0 < K; k0 += 64) {
    // Prefetch next k-tile (global_prefetch_b8).
    if (k0 + 64 < K) {
      {
        int e = tid * 32;  // one line per thread across the A tile
        int r = e >> 6, c = e & 63;
        if constexpr (A_F32)
          __builtin_prefetch(&((const float*)Av)[(size_t)(m0 + r) * K + k0 + 64 + c], 0, 3);
        else
          __builtin_prefetch(&((const __bf16*)Av)[(size_t)(m0 + r) * K + k0 + 64 + c], 0, 3);
      }
      {
        int e = tid * 32;
        int kk = e >> 7, n = e & 127;
        __builtin_prefetch(&Bw[(size_t)(k0 + 64 + kk) * N + n0 + n], 0, 3);
      }
    }
    // Stage A tile 128x64 (row-major), converting to bf16.
#pragma unroll
    for (int i = 0; i < 32; ++i) {
      int e = tid + i * 256;
      int r = e >> 6, c = e & 63;
      float a;
      if constexpr (A_F32)
        a = ((const float*)Av)[(size_t)(m0 + r) * K + k0 + c];
      else
        a = (float)((const __bf16*)Av)[(size_t)(m0 + r) * K + k0 + c];
      As[r * LD + c] = (__bf16)a;
    }
    // Stage B tile 64x128 transposed: Bs[n][k] = Bw[k0+k][n0+n].
#pragma unroll
    for (int i = 0; i < 32; ++i) {
      int e = tid + i * 256;
      int kk = e >> 7, n = e & 127;
      Bs[n * LD + kk] = (__bf16)Bw[(size_t)(k0 + kk) * N + n0 + n];
    }
    __syncthreads();

#pragma unroll
    for (int ks = 0; ks < 2; ++ks) {
      v16bf af[2], bf[4];
#pragma unroll
      for (int i = 0; i < 2; ++i)
        af[i] = load_frag_b(As, LD, 32 * wm + 16 * i + l16, half, 32 * ks);
#pragma unroll
      for (int t = 0; t < 4; ++t)
        bf[t] = load_frag_b(Bs, LD, 64 * wn + 16 * t + l16, half, 32 * ks);
#pragma unroll
      for (int i = 0; i < 2; ++i)
#pragma unroll
        for (int t = 0; t < 4; ++t)
          acc[i][t] = __builtin_amdgcn_wmma_f32_16x16x32_bf16(
              false, af[i], false, bf[t], (short)0, acc[i][t], false, false);
    }
    __syncthreads();
  }

  // Epilogue. C/D layout: VGPR r -> row r + 8*half, col = l16.
#pragma unroll
  for (int i = 0; i < 2; ++i)
#pragma unroll
    for (int t = 0; t < 4; ++t)
#pragma unroll
      for (int r = 0; r < 8; ++r) {
        int mrow = m0 + 32 * wm + 16 * i + r + 8 * half;
        int ncol = n0 + 64 * wn + 16 * t + l16;
        float v = (acc[i][t][r] + bias[ncol]) * scale;
        if constexpr (HEADED) {
          int b = mrow >> 11, s = mrow & (S_LEN - 1);
          int h = ncol >> 6, dh = ncol & (DHEAD - 1);
          ((__bf16*)Ov)[(((size_t)(b * NHEAD + h)) * S_LEN + s) * DHEAD + dh] =
              (__bf16)v;
        } else {
          ((float*)Ov)[(size_t)mrow * N + ncol] = v;
        }
      }
}

// ---------------------------------------------------------------------------
// Flash attention. Grid (S/64, H, B), block 128 (4 waves).
// Each wave owns 16 query rows; 64-key KV tiles; online softmax.
// K / Q / mask tiles staged via async-to-LDS (ASYNCcnt); V staged with a
// transpose through normal DS stores; P slab reuse guarded by s_wait_dscnt.
// ---------------------------------------------------------------------------
__global__ __launch_bounds__(128) void flash_attn_k(
    const __bf16* __restrict__ Q, const __bf16* __restrict__ Kg,
    const __bf16* __restrict__ Vg, const uint8_t* __restrict__ mask,
    __bf16* __restrict__ Cx) {
  constexpr int LDA = 72;  // 144 B rows: 16B-aligned for b128 async stores,
                           // 36-dword stride (conflict-free over 16 rows)
  constexpr int LDV = 66;
  __shared__ __bf16  Ks[64 * LDA];      // K tile, row-major [key][dh]
  __shared__ __bf16  Vt[64 * LDV];      // V tile transposed [dh][key]
  __shared__ __bf16  Ps[4][16 * LDA];   // per-wave P slab (also Q staging)
  __shared__ uint8_t Ms[64 * 64];       // mask tile [qrow][key]

  const int tid  = threadIdx.x;
  const int lane = tid & 31, wave = tid >> 5;
  const int l16 = lane & 15, half = lane >> 4;
  const int qb = blockIdx.x, h = blockIdx.y, b = blockIdx.z;
  const int bh = b * NHEAD + h;
  const int q0 = qb * 64;

  // ---- Stage 64x64 Q tile into the Ps slab (flat 64 x LDA) ----
  const __bf16* Qbase = Q + ((size_t)bh * S_LEN + q0) * DHEAD;
  __bf16* Qs = &Ps[0][0];
#ifdef HAVE_ASYNC_LDS
#pragma unroll
  for (int i = 0; i < 4; ++i) {  // 512 x 16B chunks / 128 threads
    int idx = tid + i * 128;
    int r = idx >> 3, ch = idx & 7;  // 8 chunks of 8 bf16 per 64-elem row
    async_copy_b128(Qbase + r * DHEAD + ch * 8, &Qs[r * LDA + ch * 8]);
  }
  wait_async0();
#else
  for (int e = tid; e < 64 * 32; e += 128) {
    int r = e >> 5, c = e & 31;
    *(unsigned int*)&Qs[r * LDA + 2 * c] =
        ((const unsigned int*)Qbase)[r * 32 + c];
  }
#endif
  __syncthreads();
  v16bf qf[2];
#pragma unroll
  for (int ks = 0; ks < 2; ++ks)
    qf[ks] = load_frag_b(Qs, LDA, 16 * wave + l16, half, 32 * ks);

  float m_[8], l_[8];
#pragma unroll
  for (int r = 0; r < 8; ++r) { m_[r] = -1e30f; l_[r] = 0.0f; }
  v8f acc[4] = {};

  const uint8_t* maskb = mask + (size_t)b * S_LEN * S_LEN + (size_t)q0 * S_LEN;

  for (int j = 0; j < S_LEN / 64; ++j) {
    const int kv0 = j * 64;
    __syncthreads();  // previous iter done reading Ks/Vt/Ms (and Qs frags)

    const __bf16* Kbase = Kg + ((size_t)bh * S_LEN + kv0) * DHEAD;
    const __bf16* Vbase = Vg + ((size_t)bh * S_LEN + kv0) * DHEAD;

    // Prefetch next KV tiles.
    if (kv0 + 64 < S_LEN) {
      __builtin_prefetch(Kbase + 64 * DHEAD + tid * 32, 0, 3);
      __builtin_prefetch(Vbase + 64 * DHEAD + tid * 32, 0, 3);
    }

#ifdef HAVE_ASYNC_LDS
    // K tile: 64 rows x 128 B, async b128 chunks.
#pragma unroll
    for (int i = 0; i < 4; ++i) {
      int idx = tid + i * 128;
      int r = idx >> 3, ch = idx & 7;
      async_copy_b128(Kbase + r * DHEAD + ch * 8, &Ks[r * LDA + ch * 8]);
    }
    // Mask tile: 64 rows x 64 B, async b128 chunks.
#pragma unroll
    for (int i = 0; i < 2; ++i) {
      int idx = tid + i * 128;
      int r = idx >> 2, ch = idx & 3;
      async_copy_b128(maskb + (size_t)r * S_LEN + kv0 + ch * 16,
                      &Ms[r * 64 + ch * 16]);
    }
#else
    for (int e = tid; e < 64 * 32; e += 128) {
      int r = e >> 5, c = e & 31;
      *(unsigned int*)&Ks[r * LDA + 2 * c] =
          ((const unsigned int*)Kbase)[r * 32 + c];
    }
    for (int idx = tid; idx < 256; idx += 128) {
      int r = idx >> 2, ch = idx & 3;
      *(uint4*)&Ms[r * 64 + ch * 16] =
          *(const uint4*)(maskb + (size_t)r * S_LEN + kv0 + ch * 16);
    }
#endif
    // V tile transposed (needs element shuffle -> normal DS stores).
    for (int e = tid; e < 64 * 32; e += 128) {
      int r = e >> 5, c = e & 31;  // key r, dh pair c
      unsigned int vv = ((const unsigned int*)Vbase)[r * 32 + c];
      Vt[(2 * c) * LDV + r]     = ((const __bf16*)&vv)[0];
      Vt[(2 * c + 1) * LDV + r] = ((const __bf16*)&vv)[1];
    }
#ifdef HAVE_ASYNC_LDS
    wait_async0();
#endif
    __syncthreads();

    // S = Q K^T : B-fragment lane = key row of Ks (columns of K^T).
    v8f st[4] = {};
#pragma unroll
    for (int ks = 0; ks < 2; ++ks)
#pragma unroll
      for (int t = 0; t < 4; ++t) {
        v16bf kf = load_frag_b(Ks, LDA, 16 * t + l16, half, 32 * ks);
        st[t] = __builtin_amdgcn_wmma_f32_16x16x32_bf16(
            false, qf[ks], false, kf, (short)0, st[t], false, false);
      }

    // Apply boolean mask from LDS tile (True -> -1e30).
#pragma unroll
    for (int t = 0; t < 4; ++t)
#pragma unroll
      for (int r = 0; r < 8; ++r) {
        if (Ms[(16 * wave + r + 8 * half) * 64 + 16 * t + l16])
          st[t][r] = -1e30f;
      }

    // Online softmax: row max across 4 N-tiles + 16-lane half-wave reduce.
    float mnew[8];
#pragma unroll
    for (int r = 0; r < 8; ++r)
      mnew[r] = fmaxf(m_[r], fmaxf(fmaxf(st[0][r], st[1][r]),
                                   fmaxf(st[2][r], st[3][r])));
#pragma unroll
    for (int xm = 1; xm <= 8; xm <<= 1)
#pragma unroll
      for (int r = 0; r < 8; ++r)
        mnew[r] = fmaxf(mnew[r], __shfl_xor(mnew[r], xm, 32));

    float rsum[8] = {};
#pragma unroll
    for (int t = 0; t < 4; ++t)
#pragma unroll
      for (int r = 0; r < 8; ++r) {
        float p = __expf(st[t][r] - mnew[r]);
        rsum[r] += p;
        Ps[wave][(r + 8 * half) * LDA + 16 * t + l16] = (__bf16)p;
      }
#pragma unroll
    for (int xm = 1; xm <= 8; xm <<= 1)
#pragma unroll
      for (int r = 0; r < 8; ++r) rsum[r] += __shfl_xor(rsum[r], xm, 32);

#pragma unroll
    for (int r = 0; r < 8; ++r) {
      float cs = __expf(m_[r] - mnew[r]);
      l_[r] = l_[r] * cs + rsum[r];
      m_[r] = mnew[r];
#pragma unroll
      for (int t = 0; t < 4; ++t) acc[t][r] *= cs;
    }

    // Wave-local LDS RAW: split dependency counter, no block barrier needed.
    asm volatile("s_wait_dscnt 0" ::: "memory");

    // ctx += P V : A = P (from per-wave slab), B lane = dh row of Vt.
#pragma unroll
    for (int ks = 0; ks < 2; ++ks) {
      v16bf pf = load_frag_b(&Ps[wave][0], LDA, l16, half, 32 * ks);
#pragma unroll
      for (int t = 0; t < 4; ++t) {
        v16bf vf = load_frag_b(Vt, LDV, 16 * t + l16, half, 32 * ks);
        acc[t] = __builtin_amdgcn_wmma_f32_16x16x32_bf16(
            false, pf, false, vf, (short)0, acc[t], false, false);
      }
    }
  }

  // Normalize and write ctx back in [B, S, D] layout (bf16).
#pragma unroll
  for (int r = 0; r < 8; ++r) {
    float inv = 1.0f / l_[r];
    int q = q0 + 16 * wave + r + 8 * half;
#pragma unroll
    for (int t = 0; t < 4; ++t) {
      int dh = 16 * t + l16;
      Cx[((size_t)b * S_LEN + q) * DMODEL + h * DHEAD + dh] =
          (__bf16)(acc[t][r] * inv);
    }
  }
}

// ---------------------------------------------------------------------------
extern "C" void kernel_launch(void* const* d_in, const int* in_sizes, int n_in,
                              void* d_out, int out_size, void* d_ws,
                              size_t ws_size, hipStream_t stream) {
  (void)in_sizes; (void)n_in; (void)out_size; (void)ws_size;
  const float*   key   = (const float*)d_in[0];
  const float*   value = (const float*)d_in[1];
  const float*   query = (const float*)d_in[2];
  const uint8_t* maskp = (const uint8_t*)d_in[3];
  const float*   Wq = (const float*)d_in[4];
  const float*   bq = (const float*)d_in[5];
  const float*   Wk = (const float*)d_in[6];
  const float*   bk = (const float*)d_in[7];
  const float*   Wv = (const float*)d_in[8];
  const float*   bv = (const float*)d_in[9];
  const float*   Wo = (const float*)d_in[10];
  const float*   bo = (const float*)d_in[11];

  const size_t BSD = (size_t)BATCH * S_LEN * DMODEL;  // 4M elements
  __bf16* Qb = (__bf16*)d_ws;
  __bf16* Kb = Qb + BSD;
  __bf16* Vb = Kb + BSD;
  __bf16* Cx = Vb + BSD;

  dim3 ggrid(DMODEL / 128, (BATCH * S_LEN) / 128);  // (8, 32)
  const float qscale = 0.125f;  // 1/sqrt(DH=64)

  gemm_wmma_k<true, true><<<ggrid, 256, 0, stream>>>(query, Wq, bq, Qb, qscale);
  gemm_wmma_k<true, true><<<ggrid, 256, 0, stream>>>(key,   Wk, bk, Kb, 1.0f);
  gemm_wmma_k<true, true><<<ggrid, 256, 0, stream>>>(value, Wv, bv, Vb, 1.0f);

  flash_attn_k<<<dim3(S_LEN / 64, NHEAD, BATCH), 128, 0, stream>>>(
      Qb, Kb, Vb, maskp, Cx);

  gemm_wmma_k<false, false><<<ggrid, 256, 0, stream>>>(Cx, Wo, bo,
                                                       (float*)d_out, 1.0f);
}
